// RelNetGroupAttention2_55834574848355
// MI455X (gfx1250) — matrix operand order repaired and checked by reading, not compiled, not run on hardware
//
#include <hip/hip_runtime.h>
#include <hip/hip_bf16.h>
#include <math.h>

// ---------------------------------------------------------------------------
// RelNet forward for MI455X (gfx1250, wave32, WMMA).
// All heavy GEMMs use v_wmma_f32_16x16x32_bf16 (bf16 in, fp32 accum).
// Dominant cost (3x GEMM 131072x256x256) uses MBLK=4 register blocking:
// one B fragment per 4 WMMAs. Conv2-4 are implicit-GEMM over NHWC bf16 with
// per-tap hoisted addressing (tap = (kh,kw), 64 channels = 2 K-steps).
// Workspace requirement: ~276 MB.
// ---------------------------------------------------------------------------

typedef __attribute__((ext_vector_type(16))) __bf16 v16bf;
typedef __attribute__((ext_vector_type(8)))  __bf16 v8bf;
typedef __attribute__((ext_vector_type(8)))  float  v8f;

__device__ __forceinline__ float  bf2f(__bf16 x) { return (float)x; }
__device__ __forceinline__ __bf16 f2bf(float x)  { return (__bf16)x; }
__device__ __forceinline__ float  sigm(float x)  { return 1.0f / (1.0f + __expf(-x)); }

// ---- WMMA fragment loaders (layouts per CDNA5 ISA §7.12.2) ------
// A (16x32 bf16): lane l -> row m0+(l&15); elems 0..7 = K {base..base+7},
// elems 8..15 = K {base+16..}, base = k0 + (l<16 ? 0 : 8).
__device__ __forceinline__ v16bf frag_a(const __bf16* __restrict__ A, int lda,
                                        int m0, int k0, int lane) {
  const int half = lane >> 4, mr = lane & 15;
  const __bf16* p = A + (size_t)(m0 + mr) * lda + (k0 + half * 8);
  v8bf lo = *(const v8bf*)p;
  v8bf hi = *(const v8bf*)(p + 16);
  return __builtin_shufflevector(lo, hi, 0,1,2,3,4,5,6,7,8,9,10,11,12,13,14,15);
}
// B (32x16 bf16), stored as Bt(N,K) row-major: lane l -> col n0+(l&15),
// elems 0..7 = K {base..base+7}, elems 8..15 = K {base+16..}, base = k0 + (l<16 ? 0 : 8).
__device__ __forceinline__ v16bf frag_b(const __bf16* __restrict__ Bt, int ldb,
                                        int n0, int k0, int lane) {
  const int half = lane >> 4, nr = lane & 15;
  const __bf16* p = Bt + (size_t)(n0 + nr) * ldb + (k0 + half * 8);
  v8bf lo = *(const v8bf*)p;
  v8bf hi = *(const v8bf*)(p + 16);
  return __builtin_shufflevector(lo, hi, 0,1,2,3,4,5,6,7,8,9,10,11,12,13,14,15);
}

// ---- Generic bf16 WMMA GEMM: D(M,N) = A(M,K) * Bt(N,K)^T [+bias][relu] ----
// One wave computes a (16*MBLK) x 16 strip: B fragment reused MBLK times.
template<int MBLK, bool BIAS, bool RELU, bool OUT_BF16>
__launch_bounds__(256)
__global__ void k_gemm(const __bf16* __restrict__ A, int lda,
                       const __bf16* __restrict__ Bt, int ldb,
                       const float* __restrict__ bias,
                       void* __restrict__ Out, int ldo,
                       int Mtiles, int Ntiles, int K)
{
  const int lane = threadIdx.x & 31;
  const int wave = (blockIdx.x << 3) + (threadIdx.x >> 5);
  const int Mblocks = Mtiles / MBLK;                   // Mtiles % MBLK == 0
  if (wave >= Mblocks * Ntiles) return;                // wave-uniform exit
  const int tm = wave / Ntiles, tn = wave - tm * Ntiles;
  const int m0 = (tm * MBLK) << 4, n0 = tn << 4;
  v8f acc[MBLK];
#pragma unroll
  for (int i = 0; i < MBLK; ++i) acc[i] = (v8f){};
  for (int k0 = 0; k0 < K; k0 += 32) {
    const v16bf b = frag_b(Bt, ldb, n0, k0, lane);
#pragma unroll
    for (int i = 0; i < MBLK; ++i) {
      const v16bf a = frag_a(A, lda, m0 + (i << 4), k0, lane);
      acc[i] = __builtin_amdgcn_wmma_f32_16x16x32_bf16(false, a, false, b,
                                                       (short)0, acc[i], false, false);
    }
  }
  const int half = lane >> 4, nr = lane & 15;
  const int col = n0 + nr;
  const float bs = BIAS ? bias[col] : 0.0f;
#pragma unroll
  for (int i = 0; i < MBLK; ++i) {
#pragma unroll
    for (int r = 0; r < 8; ++r) {
      float v = acc[i][r] + bs;
      if constexpr (RELU) v = fmaxf(v, 0.0f);
      const size_t idx = (size_t)(m0 + (i << 4) + r + (half << 3)) * ldo + col;
      if constexpr (OUT_BF16) ((__bf16*)Out)[idx] = f2bf(v);
      else                    ((float*)Out)[idx]  = v;
    }
  }
}

// ---- Implicit-GEMM conv 64->64, 3x3, stride 2, pad 1, NHWC bf16 -----------
// K ordered (kh,kw,c): tap (kh,kw) covers K block [tap*64, tap*64+64) ->
// exactly 2 K=32 WMMA steps reading the SAME input pixel. Address math and
// bounds check hoisted per tap; taps fully unrolled (kh/kw compile-time).
// Epilogue: folded BN (y = acc*s + t) + ReLU.
__launch_bounds__(256)
__global__ void k_conv_igemm(const __bf16* __restrict__ in, int Hi, int Wi,
                             const __bf16* __restrict__ Wc,   // (64,576) bf16
                             const float* __restrict__ bns,
                             const float* __restrict__ bnt,
                             __bf16* __restrict__ out, int Ho, int Wo, int Mtiles)
{
  const int lane = threadIdx.x & 31;
  const int wave = (blockIdx.x << 3) + (threadIdx.x >> 5);
  if (wave >= Mtiles * 4) return;                      // Ntiles = 64/16 = 4
  const int tm = wave >> 2, tn = wave & 3;
  const int m0 = tm << 4, n0 = tn << 4;
  const int half = lane >> 4, lr = lane & 15;
  const int m = m0 + lr;
  const int hw = Ho * Wo;
  const int n_img = m / hw;
  const int rem = m - n_img * hw;
  const int oh = rem / Wo, ow = rem - oh * Wo;
  const __bf16* __restrict__ wrow = Wc + (size_t)(n0 + lr) * 576 + half * 8;
  const int coff = half * 8;                           // A chunk base within tap
  v8f acc = {};
#pragma unroll
  for (int tap = 0; tap < 9; ++tap) {
    const int kh = tap / 3, kw = tap - kh * 3;         // compile-time after unroll
    const int ih = (oh << 1) - 1 + kh;
    const int iw = (ow << 1) - 1 + kw;
    const bool valid = (unsigned)ih < (unsigned)Hi && (unsigned)iw < (unsigned)Wi;
    const __bf16* __restrict__ px = in + (((size_t)n_img * Hi + ih) * Wi + iw) * 64;
#pragma unroll
    for (int c0 = 0; c0 < 64; c0 += 32) {
      v8bf lo = {}, hi = {};
      if (valid) {
        lo = *(const v8bf*)(px + c0 + coff);
        hi = *(const v8bf*)(px + c0 + 16 + coff);
      }
      const v16bf a = __builtin_shufflevector(lo, hi,
          0,1,2,3,4,5,6,7,8,9,10,11,12,13,14,15);
      const v8bf bl = *(const v8bf*)(wrow + tap * 64 + c0);
      const v8bf bh = *(const v8bf*)(wrow + tap * 64 + c0 + 16);
      const v16bf b = __builtin_shufflevector(bl, bh,
          0,1,2,3,4,5,6,7,8,9,10,11,12,13,14,15);
      acc = __builtin_amdgcn_wmma_f32_16x16x32_bf16(false, a, false, b,
                                                    (short)0, acc, false, false);
    }
  }
  const int col = n0 + lr;
  const float s = bns[col], t = bnt[col];
#pragma unroll
  for (int r = 0; r < 8; ++r) {
    float v = fmaxf(acc[r] * s + t, 0.0f);
    out[(size_t)(m0 + r + (half << 3)) * 64 + col] = f2bf(v);
  }
}

// ---- conv1: 3->64, 3x3 s2 p1, fp32 direct (K=27 too small for WMMA) -------
__global__ void k_conv1(const float* __restrict__ img, const float* __restrict__ w,
                        const float* __restrict__ bns, const float* __restrict__ bnt,
                        __bf16* __restrict__ out)
{
  const size_t id = (size_t)blockIdx.x * blockDim.x + threadIdx.x;
  if (id >= (size_t)128 * 64 * 64 * 64) return;
  const int co = (int)(id & 63);
  const int ow = (int)((id >> 6) & 63);
  const int oh = (int)((id >> 12) & 63);
  const int n  = (int)(id >> 18);
  float acc = 0.0f;
#pragma unroll
  for (int kh = 0; kh < 3; ++kh) {
    const int ih = oh * 2 - 1 + kh;
    if ((unsigned)ih >= 128u) continue;
#pragma unroll
    for (int kw = 0; kw < 3; ++kw) {
      const int iw = ow * 2 - 1 + kw;
      if ((unsigned)iw >= 128u) continue;
#pragma unroll
      for (int ci = 0; ci < 3; ++ci)
        acc += img[(((size_t)n * 3 + ci) * 128 + ih) * 128 + iw]
             * w[((co * 3 + ci) * 3 + kh) * 3 + kw];
    }
  }
  out[id] = f2bf(fmaxf(acc * bns[co] + bnt[co], 0.0f));
}

// ---- LSTM ------------------------------------------------------------------
__global__ void k_lstm_init(float* h, float* c, float* q, __bf16* hbf) {
  const int id = blockIdx.x * blockDim.x + threadIdx.x;
  if (id < 128 * 512) { h[id] = 0.0f; c[id] = 0.0f; q[id] = 0.0f; hbf[id] = f2bf(0.0f); }
}

__global__ void k_lstm_step(const float* __restrict__ XW, const float* __restrict__ HW,
                            const float* __restrict__ bih, const float* __restrict__ bhh,
                            const int* __restrict__ qlen, int t,
                            float* __restrict__ h, float* __restrict__ c,
                            __bf16* __restrict__ hbf, float* __restrict__ q)
{
  const int id = blockIdx.x * blockDim.x + threadIdx.x;
  if (id >= 128 * 512) return;
  const int n = id >> 9, j = id & 511;
  const float* xw = XW + (size_t)(n * 32 + t) * 2048;
  const float* hw = HW + (size_t)n * 2048;
  const float gi = xw[j]        + hw[j]        + bih[j]        + bhh[j];
  const float gf = xw[512 + j]  + hw[512 + j]  + bih[512 + j]  + bhh[512 + j];
  const float gg = xw[1024 + j] + hw[1024 + j] + bih[1024 + j] + bhh[1024 + j];
  const float go = xw[1536 + j] + hw[1536 + j] + bih[1536 + j] + bhh[1536 + j];
  const float cn = sigm(gf) * c[id] + sigm(gi) * tanhf(gg);
  const float hn = sigm(go) * tanhf(cn);
  c[id] = cn; h[id] = hn; hbf[id] = f2bf(hn);
  if (t == qlen[n] - 1) q[id] = hn;   // hidden at t = len-1
}

// ---- small prep / elementwise kernels -------------------------------------
__global__ void k_cvt(const float* __restrict__ s, __bf16* __restrict__ d, int n) {
  const int id = blockIdx.x * blockDim.x + threadIdx.x;
  if (id < n) d[id] = f2bf(s[id]);
}

__global__ void k_cvt_pad(const float* __restrict__ s, __bf16* __restrict__ d,
                          int rows, int ksrc, int kdst) {
  const int id = blockIdx.x * blockDim.x + threadIdx.x;
  if (id >= rows * kdst) return;
  const int r = id / kdst, k = id - r * kdst;
  d[id] = f2bf(k < ksrc ? s[(size_t)r * ksrc + k] : 0.0f);
}

__global__ void k_embed(const float* __restrict__ emb, const int* __restrict__ tok,
                        __bf16* __restrict__ d) {
  const int id = blockIdx.x * blockDim.x + threadIdx.x;
  if (id >= 4096 * 320) return;
  const int m = id / 320, k = id - m * 320;
  d[id] = f2bf(k < 300 ? emb[(size_t)tok[m] * 300 + k] : 0.0f);
}

__global__ void k_split_gw1(const float* __restrict__ gw1,
                            __bf16* __restrict__ Wa, __bf16* __restrict__ Wb,
                            __bf16* __restrict__ Wq) {
  const int id = blockIdx.x * blockDim.x + threadIdx.x;
  if (id >= 256 * 640) return;
  const int o = id / 640, c = id - o * 640;
  const __bf16 v = f2bf(gw1[id]);
  if (c < 64)       Wa[o * 64 + c] = v;
  else if (c < 128) Wb[o * 64 + (c - 64)] = v;
  else              Wq[o * 512 + (c - 128)] = v;
}

__global__ void k_convw(const float* __restrict__ cws, __bf16* __restrict__ Wc) {
  const int id = blockIdx.x * blockDim.x + threadIdx.x;
  if (id >= 3 * 64 * 64 * 9) return;        // cws: (l,o,ci,kh,kw)
  const int p  = id % 9;
  int rest = id / 9;
  const int ci = rest & 63; rest >>= 6;
  const int o  = rest & 63;
  const int l  = rest >> 6;
  Wc[((size_t)l * 64 + o) * 576 + p * 64 + ci] = f2bf(cws[id]);
}

__global__ void k_bnprep(const float* __restrict__ cbs, const float* __restrict__ bng,
                         const float* __restrict__ bnb, const float* __restrict__ bnm,
                         const float* __restrict__ bnv,
                         float* __restrict__ bns, float* __restrict__ bnt) {
  const int id = blockIdx.x * blockDim.x + threadIdx.x;
  if (id >= 4 * 64) return;
  const float s = bng[id] * rsqrtf(bnv[id] + 1e-5f);
  bns[id] = s;
  bnt[id] = (cbs[id] - bnm[id]) * s + bnb[id];   // folds conv bias + BN shift
}

// pooled[n,g,c] = obj[n,2g,c] + obj[n,2g+1,c]  (softmax over singleton == 1)
__global__ void k_pool(const __bf16* __restrict__ obj, __bf16* __restrict__ pooled) {
  const int id = blockIdx.x * blockDim.x + threadIdx.x;
  if (id >= 128 * 32 * 64) return;
  const int ch = id & 63, g = (id >> 6) & 31, n = id >> 11;
  const __bf16* p = obj + ((size_t)n * 64 + g * 2) * 64 + ch;
  pooled[id] = f2bf(bf2f(p[0]) + bf2f(p[64]));
}

// hg1[n,g1,g2,:] = relu(Pa[n,g1]+Pb[n,g2]+Qp[n]+gbs0)  (factorized g-layer 1)
__global__ void k_hg1(const float* __restrict__ Pa, const float* __restrict__ Pb,
                      const float* __restrict__ Qp, const float* __restrict__ gb,
                      __bf16* __restrict__ hg)
{
  const size_t id = (size_t)blockIdx.x * blockDim.x + threadIdx.x;
  if (id >= (size_t)131072 * 256) return;
  const int d = (int)(id & 255);
  const size_t m = id >> 8;
  const int g2 = (int)(m & 31);
  const int g1 = (int)((m >> 5) & 31);
  const int n  = (int)(m >> 10);
  const float v = Pa[(((size_t)n * 32 + g1) << 8) + d]
                + Pb[(((size_t)n * 32 + g2) << 8) + d]
                + Qp[((size_t)n << 8) + d] + gb[d];
  hg[id] = f2bf(fmaxf(v, 0.0f));
}

__global__ void k_ctx(const __bf16* __restrict__ hg, float* __restrict__ ctx,
                      __bf16* __restrict__ ctxbf) {
  const int n = blockIdx.x, d = threadIdx.x;       // 128 blocks x 256 threads
  const __bf16* p = hg + (size_t)n * 1024 * 256 + d;
  float s = 0.0f;
  for (int j = 0; j < 1024; ++j) {
    __builtin_prefetch(p + (size_t)(j + 8) * 256, 0, 1);  // global_prefetch_b8
    s += bf2f(p[(size_t)j * 256]);
  }
  s *= (1.0f / 1024.0f);
  ctx[n * 256 + d] = s;
  ctxbf[n * 256 + d] = f2bf(s);
}

__global__ void k_head(const __bf16* __restrict__ hf, const float* __restrict__ W,
                       const float* __restrict__ b, float* __restrict__ out) {
  const int n = blockIdx.x * blockDim.x + threadIdx.x;
  if (n >= 128) return;
  float l0 = b[0], l1 = b[1];
  for (int k = 0; k < 256; ++k) {
    const float h = bf2f(hf[n * 256 + k]);
    l0 += h * W[k];
    l1 += h * W[256 + k];
  }
  const float mx  = fmaxf(l0, l1);
  const float lse = mx + logf(__expf(l0 - mx) + __expf(l1 - mx));
  out[n * 2 + 0] = l0 - lse;
  out[n * 2 + 1] = l1 - lse;
}

// ---------------------------------------------------------------------------
#define DU(a, b) (((a) + (b) - 1) / (b))

extern "C" void kernel_launch(void* const* d_in, const int* in_sizes, int n_in,
                              void* d_out, int out_size, void* d_ws, size_t ws_size,
                              hipStream_t stream)
{
  (void)in_sizes; (void)n_in; (void)out_size; (void)ws_size;
  const float* img    = (const float*)d_in[0];
  const int*   qtok   = (const int*)  d_in[1];
  const int*   qlen   = (const int*)  d_in[2];
  const float* emb    = (const float*)d_in[3];
  const float* W_ih   = (const float*)d_in[4];
  const float* W_hh   = (const float*)d_in[5];
  const float* b_ih   = (const float*)d_in[6];
  const float* b_hh   = (const float*)d_in[7];
  const float* cw1    = (const float*)d_in[8];
  const float* cws    = (const float*)d_in[9];
  const float* cbs    = (const float*)d_in[10];
  const float* bng    = (const float*)d_in[11];
  const float* bnb    = (const float*)d_in[12];
  const float* bnm    = (const float*)d_in[13];
  const float* bnv    = (const float*)d_in[14];
  (void)d_in[15]; (void)d_in[16];              // attn_w / attn_b: provably unused
  const float* gw1    = (const float*)d_in[17];
  const float* gws    = (const float*)d_in[18];
  const float* gbs    = (const float*)d_in[19];
  const float* fws    = (const float*)d_in[20];
  const float* fbs    = (const float*)d_in[21];
  const float* fw_out = (const float*)d_in[22];
  const float* fb_out = (const float*)d_in[23];
  float* out = (float*)d_out;

  char* ws = (char*)d_ws;
  size_t off = 0;
  auto alloc = [&](size_t bytes) -> char* {
    char* p = ws + off;
    off += (bytes + 255) & ~(size_t)255;
    return p;
  };
  __bf16* Xq     = (__bf16*)alloc((size_t)4096 * 320 * 2);   // emb gather, K pad 300->320
  __bf16* Wih    = (__bf16*)alloc((size_t)2048 * 320 * 2);
  __bf16* Whh    = (__bf16*)alloc((size_t)2048 * 512 * 2);
  float*  XW     = (float*) alloc((size_t)4096 * 2048 * 4);  // hoisted x@W_ih^T, all t
  float*  HW     = (float*) alloc((size_t)128 * 2048 * 4);   // per-step h@W_hh^T
  float*  hbuf   = (float*) alloc((size_t)128 * 512 * 4);
  float*  cbuf   = (float*) alloc((size_t)128 * 512 * 4);
  __bf16* hbf    = (__bf16*)alloc((size_t)128 * 512 * 2);
  float*  quesv  = (float*) alloc((size_t)128 * 512 * 4);
  __bf16* quesbf = (__bf16*)alloc((size_t)128 * 512 * 2);
  __bf16* buf1   = (__bf16*)alloc((size_t)128 * 64 * 64 * 64 * 2);  // NHWC bf16
  __bf16* buf2   = (__bf16*)alloc((size_t)128 * 32 * 32 * 64 * 2);
  __bf16* buf3   = (__bf16*)alloc((size_t)128 * 16 * 16 * 64 * 2);
  __bf16* buf4   = (__bf16*)alloc((size_t)128 * 8 * 8 * 64 * 2);
  __bf16* Wc     = (__bf16*)alloc((size_t)3 * 64 * 576 * 2);
  float*  bns    = (float*) alloc((size_t)4 * 64 * 4);
  float*  bnt    = (float*) alloc((size_t)4 * 64 * 4);
  __bf16* poolbf = (__bf16*)alloc((size_t)128 * 32 * 64 * 2);
  __bf16* Wa     = (__bf16*)alloc((size_t)256 * 64 * 2);
  __bf16* Wb     = (__bf16*)alloc((size_t)256 * 64 * 2);
  __bf16* Wq     = (__bf16*)alloc((size_t)256 * 512 * 2);
  __bf16* gwsbf  = (__bf16*)alloc((size_t)3 * 256 * 256 * 2);
  __bf16* fwsbf  = (__bf16*)alloc((size_t)2 * 256 * 256 * 2);
  float*  Pa     = (float*) alloc((size_t)4096 * 256 * 4);
  float*  Pb     = (float*) alloc((size_t)4096 * 256 * 4);
  float*  Qp     = (float*) alloc((size_t)128 * 256 * 4);
  __bf16* hgA    = (__bf16*)alloc((size_t)131072 * 256 * 2);
  __bf16* hgB    = (__bf16*)alloc((size_t)131072 * 256 * 2);
  float*  ctx    = (float*) alloc((size_t)128 * 256 * 4);
  __bf16* ctxbf  = (__bf16*)alloc((size_t)128 * 256 * 2);
  __bf16* hf1    = (__bf16*)alloc((size_t)128 * 256 * 2);
  __bf16* hf2    = (__bf16*)alloc((size_t)128 * 256 * 2);

  // ---- weight prep (independent) ----
  k_cvt_pad<<<DU(2048 * 320, 256), 256, 0, stream>>>(W_ih, Wih, 2048, 300, 320);
  k_cvt<<<DU(2048 * 512, 256), 256, 0, stream>>>(W_hh, Whh, 2048 * 512);
  k_cvt<<<DU(3 * 256 * 256, 256), 256, 0, stream>>>(gws, gwsbf, 3 * 256 * 256);
  k_cvt<<<DU(2 * 256 * 256, 256), 256, 0, stream>>>(fws, fwsbf, 2 * 256 * 256);
  k_split_gw1<<<DU(256 * 640, 256), 256, 0, stream>>>(gw1, Wa, Wb, Wq);
  k_convw<<<DU(3 * 64 * 64 * 9, 256), 256, 0, stream>>>(cws, Wc);
  k_bnprep<<<1, 256, 0, stream>>>(cbs, bng, bnb, bnm, bnv, bns, bnt);
  k_embed<<<DU(4096 * 320, 256), 256, 0, stream>>>(emb, qtok, Xq);

  // ---- LSTM: hoisted input GEMM (MBLK=4), then 32 recurrent steps ----
  k_gemm<4, false, false, false><<<DU(64 * 128, 8), 256, 0, stream>>>(
      Xq, 320, Wih, 320, (const float*)nullptr, XW, 2048, 256, 128, 320);
  k_lstm_init<<<DU(128 * 512, 256), 256, 0, stream>>>(hbuf, cbuf, quesv, hbf);
  for (int t = 0; t < 32; ++t) {
    // MBLK=1: only 128 rows -> favor wave parallelism over B reuse
    k_gemm<1, false, false, false><<<DU(8 * 128, 8), 256, 0, stream>>>(
        hbf, 512, Whh, 512, (const float*)nullptr, HW, 2048, 8, 128, 512);
    k_lstm_step<<<DU(128 * 512, 256), 256, 0, stream>>>(
        XW, HW, b_ih, b_hh, qlen, t, hbuf, cbuf, hbf, quesv);
  }
  k_cvt<<<DU(128 * 512, 256), 256, 0, stream>>>(quesv, quesbf, 128 * 512);

  // ---- CNN: conv1 direct, conv2-4 implicit-GEMM WMMA (NHWC bf16) ----
  k_conv1<<<DU(128 * 64 * 64 * 64, 256), 256, 0, stream>>>(img, cw1, bns, bnt, buf1);
  k_conv_igemm<<<DU(8192 * 4, 8), 256, 0, stream>>>(
      buf1, 64, 64, Wc,            bns + 64,  bnt + 64,  buf2, 32, 32, 8192);
  k_conv_igemm<<<DU(2048 * 4, 8), 256, 0, stream>>>(
      buf2, 32, 32, Wc + 64 * 576, bns + 128, bnt + 128, buf3, 16, 16, 2048);
  k_conv_igemm<<<DU(512 * 4, 8), 256, 0, stream>>>(
      buf3, 16, 16, Wc + 2 * 64 * 576, bns + 192, bnt + 192, buf4, 8, 8, 512);

  // ---- group pool (attention == identity) + factorized first g-layer ----
  k_pool<<<DU(128 * 32 * 64, 256), 256, 0, stream>>>(buf4, poolbf);
  k_gemm<4, false, false, false><<<DU(64 * 16, 8), 256, 0, stream>>>(
      poolbf, 64, Wa, 64, (const float*)nullptr, Pa, 256, 256, 16, 64);
  k_gemm<4, false, false, false><<<DU(64 * 16, 8), 256, 0, stream>>>(
      poolbf, 64, Wb, 64, (const float*)nullptr, Pb, 256, 256, 16, 64);
  k_gemm<1, false, false, false><<<DU(8 * 16, 8), 256, 0, stream>>>(
      quesbf, 512, Wq, 512, (const float*)nullptr, Qp, 256, 8, 16, 512);
  k_hg1<<<DU((size_t)131072 * 256, 256), 256, 0, stream>>>(Pa, Pb, Qp, gbs, hgA);

  // ---- g-layers 2..4: (131072 x 256) x (256 x 256) WMMA GEMMs, MBLK=4 ----
  k_gemm<4, true, true, true><<<DU(2048 * 16, 8), 256, 0, stream>>>(
      hgA, 256, gwsbf,               256, gbs + 256, hgB, 256, 8192, 16, 256);
  k_gemm<4, true, true, true><<<DU(2048 * 16, 8), 256, 0, stream>>>(
      hgB, 256, gwsbf + 65536,       256, gbs + 512, hgA, 256, 8192, 16, 256);
  k_gemm<4, true, true, true><<<DU(2048 * 16, 8), 256, 0, stream>>>(
      hgA, 256, gwsbf + 131072,      256, gbs + 768, hgB, 256, 8192, 16, 256);

  // ---- mean over 1024 relations, f-MLP, logits + log_softmax ----
  k_ctx<<<128, 256, 0, stream>>>(hgB, ctx, ctxbf);
  k_gemm<1, true, true, true><<<DU(8 * 16, 8), 256, 0, stream>>>(
      ctxbf, 256, fwsbf,         256, fbs,       hf1, 256, 8, 16, 256);
  k_gemm<1, true, true, true><<<DU(8 * 16, 8), 256, 0, stream>>>(
      hf1,   256, fwsbf + 65536, 256, fbs + 256, hf2, 256, 8, 16, 256);
  k_head<<<1, 128, 0, stream>>>(hf2, fw_out, fb_out, out);
}
